// Attend_7834020348589
// MI455X (gfx1250) — matrix operand order repaired and checked by reading, not compiled, not run on hardware
//
#include <hip/hip_runtime.h>

#define SQ 2048
#define DH 64
#define NH 16

// 1/sqrt(64) * log2(e): scores come out of QK^T already in the exp2 domain.
#define QSCALE 0.18033688011112042f

typedef __attribute__((ext_vector_type(16))) __bf16 v16bf;
typedef __attribute__((ext_vector_type(2)))  __bf16 v2bf;
typedef __attribute__((ext_vector_type(8)))  float  v8f;
typedef __attribute__((ext_vector_type(4)))  unsigned u32x4;
typedef __attribute__((ext_vector_type(8)))  int      i32x8;
typedef __attribute__((ext_vector_type(4)))  int      i32x4;

#ifndef __has_builtin
#define __has_builtin(x) 0
#endif
#if __has_builtin(__builtin_amdgcn_tensor_load_to_lds) && \
    __has_builtin(__builtin_amdgcn_s_wait_tensorcnt)
#define HAVE_TDM 1
#else
#define HAVE_TDM 0
#endif

union U32B {
    v16bf v;
    uint4 q[2];
};

__device__ __forceinline__ unsigned packbf2(float a, float b) {
    v2bf p;
    p[0] = (__bf16)a;
    p[1] = (__bf16)b;
    return __builtin_bit_cast(unsigned, p);
}

#if HAVE_TDM
__device__ __forceinline__ unsigned lds_off(const void* p) {
    // Low 32 bits of a flat pointer into the LDS aperture are the LDS byte address.
    return (unsigned)(unsigned long long)p;
}

// Issue a TDM load of a 2D tile: `rows` rows x 64 fp32 (256B) from a row-major
// [SQ][64] fp32 tensor into LDS at byte offset lds_byte. D# per ISA 08 §8.
__device__ __forceinline__ void tdm_load_2d(const float* gaddr, unsigned lds_byte,
                                            unsigned rows) {
    const unsigned long long ga = (unsigned long long)gaddr;
    u32x4 g0;
    g0[0] = 1u;                                              // count=1 (valid), user mode
    g0[1] = lds_byte;                                        // lds_addr
    g0[2] = (unsigned)(ga & 0xFFFFFFFFu);                    // global_addr[31:0]
    g0[3] = (unsigned)((ga >> 32) & 0x1FFFFFFu) | (2u << 30);// global_addr[56:32] | type=2
    i32x8 g1;
    g1[0] = (2 << 16);        // workgroup_mask=0 | data_size=2 (4B)
    g1[1] = (64 << 16);       // tensor_dim0 = 64 elements (dw1[31:16])
    g1[2] = (2048 << 16);     // tensor_dim0 hi=0 | tensor_dim1 = 2048 (lo 16)
    g1[3] = (64 << 16);       // tensor_dim1 hi=0 | tile_dim0 = 64
    g1[4] = (int)rows;        // tile_dim1 = rows | tile_dim2 = 0
    g1[5] = 64;               // tensor_dim0_stride = 64 elements
    g1[6] = 0;                // stride0 hi | tensor_dim1_stride lo (unused for 2D)
    g1[7] = 0;
    i32x4 z4 = {0, 0, 0, 0};
#if __clang_major__ >= 23
    i32x8 z8 = {0, 0, 0, 0, 0, 0, 0, 0};
    __builtin_amdgcn_tensor_load_to_lds(g0, g1, z4, z4, z8, 0);
#else
    __builtin_amdgcn_tensor_load_to_lds(g0, g1, z4, z4, 0);
#endif
}
#endif

// Flash-attention forward, causal, bf16 WMMA with fp32 accumulation.
// Grid: H * (S/64) blocks of 128 threads (4 waves); each wave owns 16 Q rows.
__global__ __launch_bounds__(128)
void fa_fwd_kernel(const float* __restrict__ q, const float* __restrict__ k,
                   const float* __restrict__ v, float* __restrict__ out)
{
#if HAVE_TDM
    // Raw fp32 tiles delivered by the Tensor Data Mover, double-buffered.
    __shared__ __attribute__((aligned(16))) float KrawS[2][32 * 64];
    __shared__ __attribute__((aligned(16))) float VrawS[2][32 * 64];
#endif
    // K tile: [32 rows][64 d] bf16, row stride 72 bf16 (36 dw, 144B: 16B-aligned, bank-clean)
    __shared__ __attribute__((aligned(16))) unsigned KsU[32 * 36];
    // V^T tile: [64 d][32 c] bf16, row stride 40 bf16 (20 dw, 80B)
    __shared__ __attribute__((aligned(16))) unsigned VtU[64 * 20];
    // Per-wave P scratch: [16 rows][32 c] bf16, row stride 40 bf16 (20 dw)
    __shared__ __attribute__((aligned(16))) unsigned PsU[4 * 16 * 20];

    const int tid  = threadIdx.x;
    const int wave = tid >> 5;
    const int lane = tid & 31;
    const int nl   = lane & 15;   // column-within-tile / row index (A layout)
    const int hh   = lane >> 4;   // which 16-lane half

    const int blk = blockIdx.x;
    const int h   = blk >> 5;           // head
    const int i0b = (blk & 31) << 6;    // block's first query row
    const int i0w = i0b + (wave << 4);  // wave's first query row

    const float* qh = q + (size_t)h * SQ * DH;
    const float* kh = k + (size_t)h * SQ * DH;
    const float* vh = v + (size_t)h * SQ * DH;
    float*       oh = out + (size_t)h * SQ * DH;

    // ---- Q fragments: A-matrix layout for 16x16x32 bf16, two K-slabs for D=64.
    // Lane L holds row M=L%16; elems 0-7 -> d = 32t + 8*hh + i, elems 8-15 -> d = 32t + 16 + 8*hh + (i-8).
    v16bf aq[2];
    {
        const float* qr = qh + (i0w + nl) * DH;
#pragma unroll
        for (int t = 0; t < 2; ++t) {
            const float4* c0 = (const float4*)(qr + 32 * t + 8 * hh);
            const float4* c1 = (const float4*)(qr + 32 * t + 16 + 8 * hh);
            float f[16];
            *(float4*)(f + 0)  = c0[0];
            *(float4*)(f + 4)  = c0[1];
            *(float4*)(f + 8)  = c1[0];
            *(float4*)(f + 12) = c1[1];
#pragma unroll
            for (int i2 = 0; i2 < 16; ++i2)
                aq[t][i2] = (__bf16)(f[i2] * QSCALE);  // scale + log2(e) folded in
        }
    }

    v8f o0 = {}, o1 = {}, o2 = {}, o3 = {};   // O fragments for d0 = 0,16,32,48
    float m[8], l[8];
#pragma unroll
    for (int r = 0; r < 8; ++r) { m[r] = -1e30f; l[r] = 0.0f; }

    unsigned* psw = PsU + wave * (16 * 20);
    const int nsteps = (i0b >> 5) + 2;   // cover keys 0 .. i0b+63

#if HAVE_TDM
    // Prologue: DMA tile 0 (K and V) into buffer 0.
    if (wave == 0) {
        tdm_load_2d(kh, lds_off(&KrawS[0][0]), 32);
        tdm_load_2d(vh, lds_off(&VrawS[0][0]), 32);
    }
#endif

    for (int s = 0; s < nsteps; ++s) {
        const int j0 = s << 5;
#if HAVE_TDM
        const int buf = s & 1;
        if (wave == 0) {
            if (s + 1 < nsteps) {
                // Kick the DMA for the next tile, then wait for this step's two
                // transfers (TDM completes in order): allow 2 outstanding.
                const int j0n = (s + 1) << 5;
                tdm_load_2d(kh + (size_t)j0n * DH, lds_off(&KrawS[buf ^ 1][0]), 32);
                tdm_load_2d(vh + (size_t)j0n * DH, lds_off(&VrawS[buf ^ 1][0]), 32);
                __builtin_amdgcn_s_wait_tensorcnt(2);
            } else {
                __builtin_amdgcn_s_wait_tensorcnt(0);
            }
        }
        __syncthreads();   // raw tiles visible; previous step's operand reads done

        // ---- convert raw fp32 -> bf16 operand tiles (LDS -> LDS) ----
        const float* krb = &KrawS[buf][0];
        const float* vrb = &VrawS[buf][0];
#pragma unroll
        for (int it = 0; it < 8; ++it) {
            const int e = tid + it * 128;        // 1024 dword elements
            const int c = e >> 5, dp = e & 31;   // d = 2*dp
            const float2 kk = *(const float2*)(krb + c * DH + 2 * dp);
            KsU[c * 36 + dp] = packbf2(kk.x, kk.y);
        }
#pragma unroll
        for (int it = 0; it < 8; ++it) {
            const int e = tid + it * 128;
            const int d = e & 63, cp = e >> 6;   // c = 2*cp
            VtU[d * 20 + cp] = packbf2(vrb[(2 * cp) * DH + d], vrb[(2 * cp + 1) * DH + d]);
        }
#else
        __syncthreads();   // previous iteration's LDS reads done
        // ---- stage K tile (bf16, row-major) directly from global ----
#pragma unroll
        for (int it = 0; it < 8; ++it) {
            const int e = tid + it * 128;
            const int c = e >> 5, dp = e & 31;
            const float2 kk = *(const float2*)(kh + (j0 + c) * DH + 2 * dp);
            KsU[c * 36 + dp] = packbf2(kk.x, kk.y);
        }
        // ---- stage V^T tile (bf16, d-major) ----
#pragma unroll
        for (int it = 0; it < 8; ++it) {
            const int e = tid + it * 128;
            const int d = e & 63, cp = e >> 6;
            const float a0 = vh[(j0 + 2 * cp)     * DH + d];
            const float a1 = vh[(j0 + 2 * cp + 1) * DH + d];
            VtU[d * 20 + cp] = packbf2(a0, a1);
        }
        if (s + 1 < nsteps) {
            __builtin_prefetch(kh + (j0 + 32) * DH + tid * 16, 0, 0);
            __builtin_prefetch(vh + (j0 + 32) * DH + tid * 16, 0, 0);
        }
#endif
        __syncthreads();

        // ---- S = Q K^T : 16x32 scores as two 16x16 C fragments ----
        v8f c0 = {}, c1 = {};
#pragma unroll
        for (int t = 0; t < 2; ++t) {
            // B layout: lane holds key-row j0 + jn*16 + nl, elems 0-7 -> d = 32t + 8*hh + i, elems 8-15 -> d = 32t + 16 + 8*hh + (i-8)
            const unsigned* k0 = KsU + nl * 36        + 16 * t + 4 * hh;
            const unsigned* k1 = KsU + (16 + nl) * 36 + 16 * t + 4 * hh;
            U32B b0, b1;
            b0.q[0] = *(const uint4*)(k0);
            b0.q[1] = *(const uint4*)(k0 + 8);
            b1.q[0] = *(const uint4*)(k1);
            b1.q[1] = *(const uint4*)(k1 + 8);
            c0 = __builtin_amdgcn_wmma_f32_16x16x32_bf16(false, aq[t], false, b0.v,
                                                         (short)0, c0, false, false);
            c1 = __builtin_amdgcn_wmma_f32_16x16x32_bf16(false, aq[t], false, b1.v,
                                                         (short)0, c1, false, false);
        }

        // ---- online softmax (fp32, exp2 domain). Row M = r + 8*hh spans the 16 lanes of a half. ----
        const bool needmask = (j0 + 31) > i0w;   // wave-uniform
        float p0[8], p1[8];
#pragma unroll
        for (int r = 0; r < 8; ++r) {
            const int irow = i0w + r + 8 * hh;
            float x0 = c0[r];
            float x1 = c1[r];
            if (needmask) {
                if (j0 + nl      > irow) x0 = -1e30f;
                if (j0 + 16 + nl > irow) x1 = -1e30f;
            }
            float t = fmaxf(x0, x1);
            t = fmaxf(t, __shfl_xor(t, 1, 32));
            t = fmaxf(t, __shfl_xor(t, 2, 32));
            t = fmaxf(t, __shfl_xor(t, 4, 32));
            t = fmaxf(t, __shfl_xor(t, 8, 32));
            const float mn    = fmaxf(m[r], t);
            const float alpha = exp2f(m[r] - mn);   // v_exp_f32 is natively exp2
            m[r]  = mn;
            p0[r] = exp2f(x0 - mn);
            p1[r] = exp2f(x1 - mn);
            l[r]  = l[r] * alpha + p0[r] + p1[r];   // per-lane partial row sum
            o0[r] *= alpha; o1[r] *= alpha; o2[r] *= alpha; o3[r] *= alpha;
        }

        // ---- C-layout -> A-layout transpose for P via per-wave LDS ----
        {
            __bf16* ps = (__bf16*)psw;
#pragma unroll
            for (int r = 0; r < 8; ++r) {
                const int row = r + 8 * hh;
                ps[row * 40 + nl]      = (__bf16)p0[r];
                ps[row * 40 + 16 + nl] = (__bf16)p1[r];
            }
        }
        __syncthreads();

        U32B ap;
        {
            // A layout: lane row M = nl; elems 0-7 at c = 8*hh, elems 8-15 at c = 16 + 8*hh
            const unsigned* pA = psw + nl * 20 + 4 * hh;
            ap.q[0] = *(const uint4*)(pA);
            ap.q[1] = *(const uint4*)(pA + 8);
        }

        // ---- O += P * V : four 16-wide output tiles, K = 32 ----
#pragma unroll
        for (int f = 0; f < 4; ++f) {
            const unsigned* vp = VtU + (16 * f + nl) * 20 + 4 * hh;
            U32B bv;
            bv.q[0] = *(const uint4*)(vp);
            bv.q[1] = *(const uint4*)(vp + 8);
            v8f& oo = (f == 0) ? o0 : (f == 1) ? o1 : (f == 2) ? o2 : o3;
            oo = __builtin_amdgcn_wmma_f32_16x16x32_bf16(false, ap.v, false, bv.v,
                                                         (short)0, oo, false, false);
        }
    }

    // ---- epilogue: finish row sums, normalize, store fp32 ----
    float inv[8];
#pragma unroll
    for (int r = 0; r < 8; ++r) {
        float t = l[r];
        t += __shfl_xor(t, 1, 32);
        t += __shfl_xor(t, 2, 32);
        t += __shfl_xor(t, 4, 32);
        t += __shfl_xor(t, 8, 32);
        inv[r] = 1.0f / t;
    }
#pragma unroll
    for (int r = 0; r < 8; ++r) {
        float* orow = oh + (i0w + r + 8 * hh) * DH;
        orow[ 0 + nl] = o0[r] * inv[r];
        orow[16 + nl] = o1[r] * inv[r];
        orow[32 + nl] = o2[r] * inv[r];
        orow[48 + nl] = o3[r] * inv[r];
    }
}

extern "C" void kernel_launch(void* const* d_in, const int* in_sizes, int n_in,
                              void* d_out, int out_size, void* d_ws, size_t ws_size,
                              hipStream_t stream) {
    (void)in_sizes; (void)n_in; (void)d_ws; (void)ws_size; (void)out_size;
    const float* q = (const float*)d_in[0];
    const float* k = (const float*)d_in[1];
    const float* v = (const float*)d_in[2];
    float* o = (float*)d_out;
    dim3 grid(NH * (SQ / 64));   // 512 blocks
    dim3 block(128);             // 4 waves (wave32)
    hipLaunchKernelGGL(fa_fwd_kernel, grid, block, 0, stream, q, k, v, o);
}
